// SparseUNetMultiTask_8065948582438
// MI455X (gfx1250) — matrix operand order repaired and checked by reading, not compiled, not run on hardware
//
#include <hip/hip_runtime.h>
#include <hip/hip_bf16.h>

// ---------- types ----------
typedef __attribute__((ext_vector_type(16))) __bf16     v16bf;
typedef __attribute__((ext_vector_type(8)))  float      v8f;
typedef __attribute__((ext_vector_type(4)))  unsigned   u32x4;

union BfPack { u32x4 q[2]; v16bf v; };

#define KB 130            // key base (GRID+2)
#define TAB_F (130*130*130)       // 2,197,000 fine keys
#define TAB_C 1107016             // max coarse query key + 1

__device__ __forceinline__ unsigned short f2bf(float f) {
    unsigned u = __float_as_uint(f);
    unsigned r = u + 0x7fffu + ((u >> 16) & 1u);
    return (unsigned short)(r >> 16);
}

// ---------- prep kernels ----------
__global__ void fill_i32(int* p, int v, int n) {
    int i = blockIdx.x * blockDim.x + threadIdx.x;
    if (i < n) p[i] = v;
}
__global__ void fill_f32(float* p, float v, int n) {
    int i = blockIdx.x * blockDim.x + threadIdx.x;
    if (i < n) p[i] = v;
}
__global__ void build_table(const int* __restrict__ coords, int n, int* __restrict__ table) {
    int i = blockIdx.x * blockDim.x + threadIdx.x;
    if (i >= n) return;
    int k = ((coords[i*3+0] + 1) * KB + (coords[i*3+1] + 1)) * KB + (coords[i*3+2] + 1);
    table[k] = i;
}
__global__ void cvt_bf16(const float* __restrict__ s, unsigned short* __restrict__ d, int n) {
    int i = blockIdx.x * blockDim.x + threadIdx.x;
    if (i < n) d[i] = f2bf(s[i]);
}
// src: [noff][32][tiles*16] f32 row-major -> dst swizzled B-fragments:
// dst[((o*tiles+t)*32+L)*16+j] = bf16(src[o][ (j>>3)*16+(L>>4)*8+(j&7) ][ t*16+(L&15) ])
__global__ void swizzle_w(const float* __restrict__ w, unsigned short* __restrict__ d,
                          int noff, int tiles) {
    int idx = blockIdx.x * blockDim.x + threadIdx.x;
    int tot = noff * tiles * 512;
    if (idx >= tot) return;
    int j = idx & 15;
    int L = (idx >> 4) & 31;
    int t = (idx >> 9) % tiles;
    int o = (idx >> 9) / tiles;
    int k = (j >> 3) * 16 + (L >> 4) * 8 + (j & 7);
    int n = t * 16 + (L & 15);
    int cols = tiles * 16;
    d[idx] = f2bf(w[(o * 32 + k) * cols + n]);
}

// ---------- WMMA sparse conv: out = relu((sum_o gather(x,o) @ W_o) + b [FiLM]) ----------
template<int TILES, bool FILM>
__global__ __launch_bounds__(256) void spconv_wmma(
    const unsigned short* __restrict__ xbf,     // [nrows,32] bf16
    const int*            __restrict__ coords,  // [nrows,3]
    const int*            __restrict__ table, int tabSize,
    const unsigned short* __restrict__ wswz,    // [27*TILES*512] bf16 swizzled
    const float*          __restrict__ bias,    // [TILES*16]
    const unsigned short* __restrict__ condbf,  // FiLM: [nrows,32] bf16
    const unsigned short* __restrict__ filmw,   // FiLM: [4*512] bf16 swizzled
    const float*          __restrict__ filmb,   // FiLM: [64]
    float*                __restrict__ out,     // [nrows, TILES*16]
    int nrows, int ntiles)
{
    const int wave = threadIdx.x >> 5;
    const int lane = threadIdx.x & 31;
    const int tile = blockIdx.x * 8 + wave;
    if (tile >= ntiles) return;                 // wave-uniform

    const int m  = lane & 15;
    const int hi = lane >> 4;
    const int row = tile * 16 + m;
    const bool rvalid = row < nrows;

    int key;
    if (rvalid) {
        int cx = coords[row*3+0], cy = coords[row*3+1], cz = coords[row*3+2];
        key = ((cx + 1) * KB + (cy + 1)) * KB + (cz + 1);
    } else {
        key = -(1 << 30);                       // forces all lookups out of range
    }

    v8f acc[TILES];
    #pragma unroll
    for (int t = 0; t < TILES; ++t)
        for (int i = 0; i < 8; ++i) acc[t][i] = 0.f;

    for (int o = 0; o < 27; ++o) {
        const int dx = o / 9 - 1, dy = (o / 3) % 3 - 1, dz = o % 3 - 1;
        const int doff = (dx * KB + dy) * KB + dz;
        const int qk = key + doff;
        int g = -1;
        if ((unsigned)qk < (unsigned)tabSize) g = table[qk];
        const int gg = g < 0 ? 0 : g;
        // A fragment: 16-bit A layout — lane holds row m, K runs [hi*8,+8) and [16+hi*8,+8)
        const unsigned short* ap = xbf + (size_t)gg * 32 + hi * 8;
        u32x4 q0 = *(const u32x4*)(ap);
        u32x4 q1 = *(const u32x4*)(ap + 16);
        if (g < 0) {
            for (int i = 0; i < 4; ++i) { q0[i] = 0u; q1[i] = 0u; }
        }
        BfPack A; A.q[0] = q0; A.q[1] = q1;
        #pragma unroll
        for (int t = 0; t < TILES; ++t) {
            const unsigned short* wp = wswz + ((size_t)(o * TILES + t) * 32 + lane) * 16;
            BfPack Bm;
            Bm.q[0] = *(const u32x4*)(wp);
            Bm.q[1] = *(const u32x4*)(wp + 8);
            acc[t] = __builtin_amdgcn_wmma_f32_16x16x32_bf16(
                false, A.v, false, Bm.v, (short)0, acc[t], false, false);
        }
    }

    if constexpr (FILM) {
        // gamma/beta = cond @ film_w (+ film_b): 4 WMMAs, same C layout as acc
        const unsigned short* cp = condbf + (size_t)row * 32 + hi * 8;
        BfPack Ac;
        Ac.q[0] = *(const u32x4*)(cp);
        Ac.q[1] = *(const u32x4*)(cp + 16);
        v8f gb[4];
        #pragma unroll
        for (int ft = 0; ft < 4; ++ft) {
            v8f z; for (int i = 0; i < 8; ++i) z[i] = 0.f;
            const unsigned short* wp = filmw + ((size_t)ft * 32 + lane) * 16;
            BfPack Bm;
            Bm.q[0] = *(const u32x4*)(wp);
            Bm.q[1] = *(const u32x4*)(wp + 8);
            gb[ft] = __builtin_amdgcn_wmma_f32_16x16x32_bf16(
                false, Ac.v, false, Bm.v, (short)0, z, false, false);
        }
        #pragma unroll
        for (int t = 0; t < 2; ++t) {
            const int n = t * 16 + m;
            const float cb   = bias[n];
            const float gadd = filmb[n];
            const float badd = filmb[32 + n];
            #pragma unroll
            for (int r = 0; r < 8; ++r) {
                const int orow = tile * 16 + hi * 8 + r;
                float conv  = acc[t][r] + cb;
                float gamma = gb[t][r] + gadd;
                float beta  = gb[t + 2][r] + badd;
                float v = conv * (1.f + gamma) + beta;
                v = v > 0.f ? v : 0.f;
                if (orow < nrows) out[(size_t)orow * (TILES * 16) + n] = v;
            }
        }
    } else {
        #pragma unroll
        for (int t = 0; t < TILES; ++t) {
            const int n = t * 16 + m;
            const float cb = bias[n];
            #pragma unroll
            for (int r = 0; r < 8; ++r) {
                const int orow = tile * 16 + hi * 8 + r;
                float v = acc[t][r] + cb;
                v = v > 0.f ? v : 0.f;
                if (orow < nrows) out[(size_t)orow * (TILES * 16) + n] = v;
            }
        }
    }
}

// ---------- pooling / up-cond / heads ----------
__global__ void pool_add(const float* __restrict__ x1, const int* __restrict__ invc,
                         float* __restrict__ xm, float* __restrict__ cnt, int n) {
    int t = blockIdx.x * blockDim.x + threadIdx.x;
    if (t >= n * 32) return;
    int i = t >> 5, c = t & 31;
    atomicAdd(&xm[(size_t)invc[i] * 32 + c], x1[t]);
    if (c == 0) atomicAdd(&cnt[invc[i]], 1.0f);
}
__global__ void pool_div_cvt(const float* __restrict__ xm, const float* __restrict__ cnt,
                             unsigned short* __restrict__ xmbf, int nc) {
    int t = blockIdx.x * blockDim.x + threadIdx.x;
    if (t >= nc * 32) return;
    float d = cnt[t >> 5]; d = d < 1.f ? 1.f : d;
    xmbf[t] = f2bf(xm[t] / d);
}
__global__ void upcond_kernel(const float* __restrict__ x2, const int* __restrict__ invc,
                              const float* __restrict__ x1,
                              const float* __restrict__ Wup, const float* __restrict__ bup,
                              const float* __restrict__ Wc,  const float* __restrict__ bc,
                              unsigned short* __restrict__ xinbf,
                              unsigned short* __restrict__ condbf, int n) {
    int t = blockIdx.x * blockDim.x + threadIdx.x;
    if (t >= n * 32) return;
    int i = t >> 5, c = t & 31;
    const float* xb = x2 + (size_t)invc[i] * 64;
    float su = bup[c], sc = bc[c];
    #pragma unroll 4
    for (int k = 0; k < 64; ++k) {
        float xv = xb[k];
        su += xv * Wup[k * 32 + c];
        sc += xv * Wc[k * 32 + c];
    }
    xinbf[t]  = f2bf(x1[t] + su);
    condbf[t] = f2bf(sc);
}
__global__ void sem_head(const float* __restrict__ xr, const float* __restrict__ Wsem,
                         const float* __restrict__ bsem, float* __restrict__ out, int n) {
    int t = blockIdx.x * blockDim.x + threadIdx.x;
    if (t >= n * 8) return;
    int i = t >> 3, s = t & 7;
    const float* xp = xr + (size_t)i * 32;
    float acc = bsem[s];
    #pragma unroll
    for (int c = 0; c < 32; ++c) acc += xp[c] * Wsem[c * 8 + s];
    out[t] = acc;
}
__global__ void pooled_reduce(const float* __restrict__ xr, float* __restrict__ pooled, int n) {
    int t = blockIdx.x * blockDim.x + threadIdx.x;
    int c = t & 31;
    int r0 = t >> 5;
    int stride = (gridDim.x * blockDim.x) >> 5;
    float s = 0.f;
    for (int r = r0; r < n; r += stride) s += xr[(size_t)r * 32 + c];
    atomicAdd(&pooled[c], s);
}
__global__ void final_heads(const float* __restrict__ pooled, int n,
                            const float* __restrict__ Wcls, const float* __restrict__ bcls,
                            const float* __restrict__ Wemb, const float* __restrict__ bemb,
                            const float* __restrict__ Wkp,  const float* __restrict__ bkp,
                            float* __restrict__ out_cls, float* __restrict__ out_emb,
                            float* __restrict__ out_kp) {
    __shared__ float p[32];
    __shared__ float e[32];
    __shared__ float nrm;
    int t = threadIdx.x;
    if (t < 32) p[t] = pooled[t] / (float)n;
    __syncthreads();
    if (t < 4) {
        float s = bcls[t];
        for (int c = 0; c < 32; ++c) s += p[c] * Wcls[c * 4 + t];
        out_cls[t] = s;
    }
    if (t < 32) {
        float s = bemb[t];
        for (int k = 0; k < 32; ++k) s += p[k] * Wemb[k * 32 + t];
        e[t] = s;
    }
    __syncthreads();
    if (t == 0) {
        float s = 0.f;
        for (int c = 0; c < 32; ++c) s += e[c] * e[c];
        nrm = rsqrtf(s);
    }
    __syncthreads();
    if (t < 32) out_emb[t] = e[t] * nrm;
    if (t < 18) {
        int j = t / 3, d = t % 3;
        float s = bkp[t];
        for (int c = 0; c < 32; ++c) s += p[c] * Wkp[(j * 32 + c) * 3 + d];
        out_kp[t] = s;
    }
}

// ---------- host launcher ----------
extern "C" void kernel_launch(void* const* d_in, const int* in_sizes, int n_in,
                              void* d_out, int out_size, void* d_ws, size_t ws_size,
                              hipStream_t stream) {
    const int N  = in_sizes[0] / 3;
    const int Nc = in_sizes[4] / 3;

    const int*   coords = (const int*)  d_in[0];
    const float* feats  = (const float*)d_in[1];
    const int*   uc     = (const int*)  d_in[4];
    const int*   invc   = (const int*)  d_in[5];
    const float* W1     = (const float*)d_in[8];
    const float* b1     = (const float*)d_in[9];
    const float* W2     = (const float*)d_in[10];
    const float* b2     = (const float*)d_in[11];
    const float* Wup    = (const float*)d_in[12];
    const float* bup    = (const float*)d_in[13];
    const float* Wcond  = (const float*)d_in[14];
    const float* bcond  = (const float*)d_in[15];
    const float* Wf     = (const float*)d_in[16];
    const float* bfv    = (const float*)d_in[17];
    const float* film_w = (const float*)d_in[18];
    const float* film_b = (const float*)d_in[19];
    const float* Wsem   = (const float*)d_in[20];
    const float* bsem   = (const float*)d_in[21];
    const float* Wcls   = (const float*)d_in[22];
    const float* bcls   = (const float*)d_in[23];
    const float* Wemb   = (const float*)d_in[24];
    const float* bemb   = (const float*)d_in[25];
    const float* Wkp    = (const float*)d_in[26];
    const float* bkp    = (const float*)d_in[27];

    char* ws = (char*)d_ws;
    size_t off = 0;
    auto alloc = [&](size_t bytes) -> char* {
        char* p = ws + off;
        off = (off + bytes + 255) & ~(size_t)255;
        return p;
    };

    int*            tabF   = (int*)           alloc((size_t)TAB_F * 4);
    int*            tabC   = (int*)           alloc((size_t)TAB_C * 4);
    unsigned short* xbf    = (unsigned short*)alloc((size_t)N * 32 * 2);  // feats_bf, reused as xin_bf
    unsigned short* condbf = (unsigned short*)alloc((size_t)N * 32 * 2);
    float*          x1     = (float*)         alloc((size_t)N * 32 * 4);
    float*          xm     = (float*)         alloc((size_t)Nc * 32 * 4);
    float*          cnt    = (float*)         alloc((size_t)Nc * 4);
    unsigned short* xmbf   = (unsigned short*)alloc((size_t)Nc * 32 * 2);
    float*          x2     = (float*)         alloc((size_t)Nc * 64 * 4);
    float*          xr     = (float*)         alloc((size_t)N * 32 * 4);
    unsigned short* w1s    = (unsigned short*)alloc(27 * 2 * 512 * 2);
    unsigned short* w2s    = (unsigned short*)alloc(27 * 4 * 512 * 2);
    unsigned short* wfs    = (unsigned short*)alloc(27 * 2 * 512 * 2);
    unsigned short* wfilm  = (unsigned short*)alloc(4 * 512 * 2);
    float*          pooled = (float*)         alloc(32 * 4);

    float* outf = (float*)d_out;
    auto grid = [](int n, int b) { return (n + b - 1) / b; };

    // ---- prep: dense key->row tables, bf16 activations, swizzled weights ----
    fill_i32<<<grid(TAB_F, 256), 256, 0, stream>>>(tabF, -1, TAB_F);
    fill_i32<<<grid(TAB_C, 256), 256, 0, stream>>>(tabC, -1, TAB_C);
    build_table<<<grid(N, 256), 256, 0, stream>>>(coords, N, tabF);
    build_table<<<grid(Nc, 256), 256, 0, stream>>>(uc, Nc, tabC);
    cvt_bf16<<<grid(N * 32, 256), 256, 0, stream>>>(feats, xbf, N * 32);
    swizzle_w<<<grid(27 * 2 * 512, 256), 256, 0, stream>>>(W1, w1s, 27, 2);
    swizzle_w<<<grid(27 * 4 * 512, 256), 256, 0, stream>>>(W2, w2s, 27, 4);
    swizzle_w<<<grid(27 * 2 * 512, 256), 256, 0, stream>>>(Wf, wfs, 27, 2);
    swizzle_w<<<grid(4 * 512, 256), 256, 0, stream>>>(film_w, wfilm, 1, 4);

    // ---- conv1: x1 = relu(spconv(feats, W1) + b1) ----
    const int ntF = (N + 15) / 16;
    spconv_wmma<2, false><<<grid(ntF, 8), 256, 0, stream>>>(
        xbf, coords, tabF, TAB_F, w1s, b1, nullptr, nullptr, nullptr, x1, N, ntF);

    // ---- scatter-mean pool to coarse ----
    fill_f32<<<grid(Nc * 32, 256), 256, 0, stream>>>(xm, 0.f, Nc * 32);
    fill_f32<<<grid(Nc, 256), 256, 0, stream>>>(cnt, 0.f, Nc);
    pool_add<<<grid(N * 32, 256), 256, 0, stream>>>(x1, invc, xm, cnt, N);
    pool_div_cvt<<<grid(Nc * 32, 256), 256, 0, stream>>>(xm, cnt, xmbf, Nc);

    // ---- conv2 (coarse): x2 = relu(spconv(xm, W2) + b2) ----
    const int ntC = (Nc + 15) / 16;
    spconv_wmma<4, false><<<grid(ntC, 8), 256, 0, stream>>>(
        xmbf, uc, tabC, TAB_C, w2s, b2, nullptr, nullptr, nullptr, x2, Nc, ntC);

    // ---- up/cond: xin_bf = bf16(x1 + x2[invc]@Wup + bup); cond_bf = bf16(x2[invc]@Wcond + bcond)
    upcond_kernel<<<grid(N * 32, 256), 256, 0, stream>>>(
        x2, invc, x1, Wup, bup, Wcond, bcond, xbf, condbf, N);

    // ---- convf with fused FiLM: xr ----
    spconv_wmma<2, true><<<grid(ntF, 8), 256, 0, stream>>>(
        xbf, coords, tabF, TAB_F, wfs, bfv, condbf, wfilm, film_b, xr, N, ntF);

    // ---- heads ----
    sem_head<<<grid(N * 8, 256), 256, 0, stream>>>(xr, Wsem, bsem, outf, N);
    fill_f32<<<1, 32, 0, stream>>>(pooled, 0.f, 32);
    pooled_reduce<<<256, 256, 0, stream>>>(xr, pooled, N);
    final_heads<<<1, 64, 0, stream>>>(pooled, N, Wcls, bcls, Wemb, bemb, Wkp, bkp,
                                      outf + (size_t)N * 8,
                                      outf + (size_t)N * 8 + 4,
                                      outf + (size_t)N * 8 + 36);
}
